// CCDecoder_10359461118540
// MI455X (gfx1250) — matrix operand order repaired and checked by reading, not compiled, not run on hardware
//
#include <hip/hip_runtime.h>
#include <hip/hip_bf16.h>
#include <math.h>

// Problem constants (match reference)
#define B_ 256
#define S_ 512
#define T_ 16
#define H_ 1024
#define E_ 512
#define K1_ (E_ + H_)    // 1536 : E+H for W_ih

typedef __bf16 bf16_t;
typedef bf16_t v16bf __attribute__((ext_vector_type(16)));
typedef float  v8f   __attribute__((ext_vector_type(8)));

struct U32x8 { uint4 lo, hi; };

__device__ __forceinline__ v16bf make_frag(uint4 lo, uint4 hi) {
  U32x8 u; u.lo = lo; u.hi = hi;
  return __builtin_bit_cast(v16bf, u);
}

// fp32 -> bf16 round-to-nearest-even, as raw u16
__device__ __forceinline__ unsigned int f2bf(float x) {
  unsigned int u = __float_as_uint(x);
  return (u + 0x7FFFu + ((u >> 16) & 1u)) >> 16;
}

__device__ __forceinline__ float bf_lo(unsigned int u) { return __uint_as_float(u << 16); }
__device__ __forceinline__ float bf_hi(unsigned int u) { return __uint_as_float(u & 0xFFFF0000u); }

__device__ __forceinline__ float sigmoidf_(float x) { return 1.f / (1.f + __expf(-x)); }

// ---------------------------------------------------------------------------
// Gates GEMM region with compile-time strides: all loads become
// base + immediate offset (K*2 bytes <= 2KB < 24-bit IOFFSET).
// A frag (lane L, m=L&15, hi=L>>4): halves 0..7 = K[hi*8..+7], 8..15 = K[hi*8+16..+23]
// B frag: lane n holds K[hi*8..+7], K[hi*8+16..+23] of W row (gate*H + n_base + n).
template <int LDA, int LDW, int KLEN>
__device__ __forceinline__ void gates_region(
    const unsigned short* __restrict__ A,  // row m_base, k-window start already applied
    const unsigned short* __restrict__ W,  // row n_base (gate 0), k-window start applied
    int lane, v8f* acc)
{
  const int m = lane & 15, hi = lane >> 4;
  const unsigned short* pa = A + (size_t)m * LDA + hi * 8;
  const unsigned short* pb0 = W + (size_t)((0 * H_) + m) * LDW + hi * 8;
  const unsigned short* pb1 = W + (size_t)((1 * H_) + m) * LDW + hi * 8;
  const unsigned short* pb2 = W + (size_t)((2 * H_) + m) * LDW + hi * 8;
  const unsigned short* pb3 = W + (size_t)((3 * H_) + m) * LDW + hi * 8;
#pragma unroll 4
  for (int k = 0; k < KLEN; k += 32) {
    v16bf a = make_frag(*(const uint4*)(pa + k), *(const uint4*)(pa + k + 16));
    v16bf b0 = make_frag(*(const uint4*)(pb0 + k), *(const uint4*)(pb0 + k + 16));
    acc[0] = __builtin_amdgcn_wmma_f32_16x16x32_bf16(false, a, false, b0, (short)0, acc[0], false, false);
    v16bf b1 = make_frag(*(const uint4*)(pb1 + k), *(const uint4*)(pb1 + k + 16));
    acc[1] = __builtin_amdgcn_wmma_f32_16x16x32_bf16(false, a, false, b1, (short)0, acc[1], false, false);
    v16bf b2 = make_frag(*(const uint4*)(pb2 + k), *(const uint4*)(pb2 + k + 16));
    acc[2] = __builtin_amdgcn_wmma_f32_16x16x32_bf16(false, a, false, b2, (short)0, acc[2], false, false);
    v16bf b3 = make_frag(*(const uint4*)(pb3 + k), *(const uint4*)(pb3 + k + 16));
    acc[3] = __builtin_amdgcn_wmma_f32_16x16x32_bf16(false, a, false, b3, (short)0, acc[3], false, false);
  }
}

// ---------------------------------------------------------------------------
// Fused gates GEMM + LSTM pointwise. Each wave computes the 4 gate tiles
// (i,f,g,o) for one 16x16 (batch x hidden) patch, then the LSTM update.
__global__ void __launch_bounds__(128)
gates_lstm_kernel(const unsigned short* __restrict__ emb_bf,     // (B,T,E)
                  int t,
                  const unsigned short* __restrict__ htilde_bf,  // (B,H)
                  const unsigned short* __restrict__ concat_prev,// (B,2048), h at +1024
                  const unsigned short* __restrict__ Wih_bf,     // (4H,1536)
                  const unsigned short* __restrict__ Whh_bf,     // (4H,1024)
                  const float* __restrict__ b_ih,
                  const float* __restrict__ b_hh,
                  float* __restrict__ c_st,                      // (B,H)
                  float* __restrict__ h_st,                      // (B,H)
                  unsigned short* __restrict__ concat_next)      // (B,2048), h at +1024
{
  const int wave = threadIdx.x >> 5;
  const int lane = threadIdx.x & 31;
  const int m_base = blockIdx.x * 16;
  const int n_base = (blockIdx.y * 4 + wave) * 16;

  v8f acc[4] = {};

  // region 1: k in [0,512)    A = emb[:, t, :]         B = W_ih[:, 0:512)
  gates_region<T_ * E_, K1_, E_>(
      emb_bf + ((size_t)m_base * T_ + t) * E_,
      Wih_bf + (size_t)n_base * K1_, lane, acc);
  // region 2: k in [512,1536) A = h_tilde_{t-1}        B = W_ih[:, 512:1536)
  gates_region<H_, K1_, H_>(
      htilde_bf + (size_t)m_base * H_,
      Wih_bf + (size_t)n_base * K1_ + E_, lane, acc);
  // region 3: k in [0,1024)   A = h_{t-1} (concat+1024) B = W_hh
  gates_region<2048, H_, H_>(
      concat_prev + (size_t)m_base * 2048 + 1024,
      Whh_bf + (size_t)n_base * H_, lane, acc);

  // LSTM epilogue. C/D layout: lane L -> col n_base+(L&15); VGPR v -> row v+8*(L>>4).
  const int ncol = n_base + (lane & 15);
  const int hi = lane >> 4;
  const float bi = b_ih[0 * H_ + ncol] + b_hh[0 * H_ + ncol];
  const float bf = b_ih[1 * H_ + ncol] + b_hh[1 * H_ + ncol];
  const float bg = b_ih[2 * H_ + ncol] + b_hh[2 * H_ + ncol];
  const float bo = b_ih[3 * H_ + ncol] + b_hh[3 * H_ + ncol];
#pragma unroll
  for (int v = 0; v < 8; ++v) {
    const int row = m_base + v + 8 * hi;
    const size_t idx = (size_t)row * H_ + ncol;
    float iv = sigmoidf_(acc[0][v] + bi);
    float fv = sigmoidf_(acc[1][v] + bf);
    float gv = tanhf(acc[2][v] + bg);
    float ov = sigmoidf_(acc[3][v] + bo);
    float cn = fv * c_st[idx] + iv * gv;
    float hn = ov * tanhf(cn);
    c_st[idx] = cn;
    h_st[idx] = hn;
    concat_next[(size_t)row * 2048 + 1024 + ncol] = (unsigned short)f2bf(hn);
  }
}

// ---------------------------------------------------------------------------
// bf16 WMMA GEMM with compile-time LDA/K: C(256 x 1024) = A(256 x K) * W(1024 x K)^T
// MODE 0: store fp32 (attention target). MODE 1: tanh; fp32 (d_out) + bf16 (h_tilde).
template <int LDA, int K, int MODE>
__global__ void __launch_bounds__(128)
gemm_tile_kernel(const unsigned short* __restrict__ Abf,
                 const unsigned short* __restrict__ Wbf,
                 float* __restrict__ outf, int ld_out,
                 unsigned short* __restrict__ outbf)
{
  const int wave = threadIdx.x >> 5;
  const int lane = threadIdx.x & 31;
  const int m_base = blockIdx.x * 16;
  const int n_base = (blockIdx.y * 4 + wave) * 16;
  const int m = lane & 15, hi = lane >> 4;

  const unsigned short* pa = Abf + ((size_t)m_base + m) * LDA + hi * 8;
  const unsigned short* pb = Wbf + ((size_t)n_base + m) * K + hi * 8;

  v8f acc = {};
#pragma unroll 8
  for (int k = 0; k < K; k += 32) {
    v16bf a = make_frag(*(const uint4*)(pa + k), *(const uint4*)(pa + k + 16));
    v16bf b = make_frag(*(const uint4*)(pb + k), *(const uint4*)(pb + k + 16));
    acc = __builtin_amdgcn_wmma_f32_16x16x32_bf16(false, a, false, b, (short)0, acc, false, false);
  }
  const int ncol = n_base + m;
#pragma unroll
  for (int v = 0; v < 8; ++v) {
    const int row = m_base + v + 8 * hi;
    float x = acc[v];
    if (MODE == 1) {
      x = tanhf(x);
      outbf[(size_t)row * H_ + ncol] = (unsigned short)f2bf(x);
    }
    outf[(size_t)row * ld_out + ncol] = x;
  }
}

// ---------------------------------------------------------------------------
// Single-pass online-softmax attention over bf16 ctx. One block per batch,
// 128 threads; thread t owns 8 columns (one b128 load per ctx row). Each
// 16-row chunk is read ONCE into registers and reused for scores + weighted
// accumulation. ctx traffic: 256 MB/step.
__global__ void __launch_bounds__(128)
attn_bf16_kernel(const unsigned short* __restrict__ ctxb,  // (B,S,H) bf16
                 const float* __restrict__ target,         // (B,H)
                 const int* __restrict__ src_len,          // (B,)
                 unsigned short* __restrict__ concat_next) // (B,2048), weighted at +0
{
  const int b = blockIdx.x;
  const int tid = threadIdx.x;               // 0..127
  const int lane = tid & 31, wave = tid >> 5;
  const int c0 = 8 * tid;

  __shared__ float s_part[4 * 16];
  __shared__ float s_sc[16];

  float tgt[8];
#pragma unroll
  for (int j = 0; j < 8; ++j) tgt[j] = target[(size_t)b * H_ + c0 + j];

  float acc[8] = {};
  float m_run = -3.0e38f, l_run = 0.f;
  const int slen = src_len[b];
  const unsigned short* cbase = ctxb + (size_t)b * S_ * H_ + c0;

  for (int s0 = 0; s0 < S_; s0 += 16) {
    uint4 v[16];
#pragma unroll
    for (int r = 0; r < 16; ++r)
      v[r] = *(const uint4*)(cbase + (size_t)(s0 + r) * H_);

    // per-row dot products; reduce across wave then across the 4 waves in LDS
    float p[16];
#pragma unroll
    for (int r = 0; r < 16; ++r) {
      const unsigned* u = (const unsigned*)&v[r];
      float d = 0.f;
#pragma unroll
      for (int j = 0; j < 4; ++j)
        d += bf_lo(u[j]) * tgt[2 * j] + bf_hi(u[j]) * tgt[2 * j + 1];
      d += __shfl_xor(d, 16, 32);
      d += __shfl_xor(d, 8, 32);
      d += __shfl_xor(d, 4, 32);
      d += __shfl_xor(d, 2, 32);
      d += __shfl_xor(d, 1, 32);
      p[r] = d;
    }
    if (lane == 0) {
#pragma unroll
      for (int r = 0; r < 16; ++r) s_part[wave * 16 + r] = p[r];
    }
    __syncthreads();
    if (tid < 16) {
      float s = s_part[tid] + s_part[16 + tid] + s_part[32 + tid] + s_part[48 + tid];
      s_sc[tid] = (s0 + tid < slen) ? s : -1e9f;  // length masking
    }
    __syncthreads();

    float mc = -3.0e38f;
#pragma unroll
    for (int r = 0; r < 16; ++r) mc = fmaxf(mc, s_sc[r]);
    const float m_new = fmaxf(m_run, mc);
    const float scale = __expf(m_run - m_new);   // underflows to 0 on first chunk
    l_run *= scale;
#pragma unroll
    for (int j = 0; j < 8; ++j) acc[j] *= scale;
#pragma unroll
    for (int r = 0; r < 16; ++r) {
      const float pe = __expf(s_sc[r] - m_new);
      l_run += pe;
      const unsigned* u = (const unsigned*)&v[r];
#pragma unroll
      for (int j = 0; j < 4; ++j) {
        acc[2 * j]     += pe * bf_lo(u[j]);
        acc[2 * j + 1] += pe * bf_hi(u[j]);
      }
    }
    m_run = m_new;
    __syncthreads();
  }

  const float inv = 1.f / l_run;
  uint4 pk;
  unsigned* po = (unsigned*)&pk;
#pragma unroll
  for (int j = 0; j < 4; ++j)
    po[j] = f2bf(acc[2 * j] * inv) | (f2bf(acc[2 * j + 1] * inv) << 16);
  *(uint4*)(concat_next + (size_t)b * 2048 + c0) = pk;
}

// ---------------------------------------------------------------------------
// fp32-ctx fallback (used when workspace is too small for bf16 ctx copy).
__global__ void __launch_bounds__(256)
attn_f32_kernel(const float* __restrict__ ctx,
                const float* __restrict__ target,
                const int* __restrict__ src_len,
                unsigned short* __restrict__ concat_next)
{
  const int b = blockIdx.x;
  const int tid = threadIdx.x;
  const int lane = tid & 31, wave = tid >> 5;
  const int c0 = 4 * tid;

  __shared__ float s_part[8 * 16];
  __shared__ float s_sc[16];

  const float4 tgt = *(const float4*)(target + (size_t)b * H_ + c0);
  float4 acc = make_float4(0.f, 0.f, 0.f, 0.f);
  float m_run = -3.0e38f, l_run = 0.f;
  const int slen = src_len[b];
  const float* cbase = ctx + (size_t)b * S_ * H_ + c0;

  for (int s0 = 0; s0 < S_; s0 += 16) {
    float4 v[16];
#pragma unroll
    for (int r = 0; r < 16; ++r)
      v[r] = *(const float4*)(cbase + (size_t)(s0 + r) * H_);
    float p[16];
#pragma unroll
    for (int r = 0; r < 16; ++r) {
      float d = v[r].x * tgt.x + v[r].y * tgt.y + v[r].z * tgt.z + v[r].w * tgt.w;
      d += __shfl_xor(d, 16, 32);
      d += __shfl_xor(d, 8, 32);
      d += __shfl_xor(d, 4, 32);
      d += __shfl_xor(d, 2, 32);
      d += __shfl_xor(d, 1, 32);
      p[r] = d;
    }
    if (lane == 0) {
#pragma unroll
      for (int r = 0; r < 16; ++r) s_part[wave * 16 + r] = p[r];
    }
    __syncthreads();
    if (tid < 16) {
      float s = 0.f;
#pragma unroll
      for (int w = 0; w < 8; ++w) s += s_part[w * 16 + tid];
      s_sc[tid] = (s0 + tid < slen) ? s : -1e9f;
    }
    __syncthreads();
    float mc = -3.0e38f;
#pragma unroll
    for (int r = 0; r < 16; ++r) mc = fmaxf(mc, s_sc[r]);
    const float m_new = fmaxf(m_run, mc);
    const float scale = __expf(m_run - m_new);
    l_run *= scale;
    acc.x *= scale; acc.y *= scale; acc.z *= scale; acc.w *= scale;
#pragma unroll
    for (int r = 0; r < 16; ++r) {
      const float pe = __expf(s_sc[r] - m_new);
      l_run += pe;
      acc.x += pe * v[r].x; acc.y += pe * v[r].y;
      acc.z += pe * v[r].z; acc.w += pe * v[r].w;
    }
    m_run = m_new;
    __syncthreads();
  }
  const float inv = 1.f / l_run;
  uint2 pk;
  pk.x = f2bf(acc.x * inv) | (f2bf(acc.y * inv) << 16);
  pk.y = f2bf(acc.z * inv) | (f2bf(acc.w * inv) << 16);
  *(uint2*)(concat_next + (size_t)b * 2048 + c0) = pk;
}

// ---------------------------------------------------------------------------
__global__ void f32x4_to_bf16x4_kernel(const float4* __restrict__ in,
                                       uint2* __restrict__ out, int n4) {
  int i = blockIdx.x * blockDim.x + threadIdx.x;
  if (i < n4) {
    float4 f = in[i];
    uint2 o;
    o.x = f2bf(f.x) | (f2bf(f.y) << 16);
    o.y = f2bf(f.z) | (f2bf(f.w) << 16);
    out[i] = o;
  }
}

__global__ void init_state_kernel(const float* __restrict__ h0,
                                  const float* __restrict__ c0,
                                  float* __restrict__ h_st, float* __restrict__ c_st,
                                  unsigned short* __restrict__ htilde_bf,
                                  unsigned short* __restrict__ concat0) {
  int i = blockIdx.x * blockDim.x + threadIdx.x;
  if (i < B_ * H_) {
    float h = h0[i], c = c0[i];
    h_st[i] = h; c_st[i] = c;
    htilde_bf[i] = (unsigned short)f2bf(h);
    int row = i / H_, col = i % H_;
    concat0[(size_t)row * 2048 + 1024 + col] = (unsigned short)f2bf(h);
  }
}

__global__ void finalize_kernel(const float* __restrict__ h_st,
                                const float* __restrict__ c_st,
                                float* __restrict__ out_tail) {
  int i = blockIdx.x * blockDim.x + threadIdx.x;
  if (i < B_ * H_) {
    out_tail[i] = h_st[i];
    out_tail[B_ * H_ + i] = c_st[i];
  }
}

// ---------------------------------------------------------------------------
extern "C" void kernel_launch(void* const* d_in, const int* in_sizes, int n_in,
                              void* d_out, int out_size, void* d_ws, size_t ws_size,
                              hipStream_t stream) {
  const float* trg_emb = (const float*)d_in[0];   // (B,T,E)
  const float* h0      = (const float*)d_in[1];   // (B,H)
  const float* c0      = (const float*)d_in[2];   // (B,H)
  const float* ctx     = (const float*)d_in[3];   // (B,S,H)
  const int*   src_len = (const int*)d_in[4];     // (B,)
  const float* W_ih    = (const float*)d_in[5];   // (4H, E+H)
  const float* W_hh    = (const float*)d_in[6];   // (4H, H)
  const float* b_ih    = (const float*)d_in[7];   // (4H,)
  const float* b_hh    = (const float*)d_in[8];   // (4H,)
  const float* W_in    = (const float*)d_in[9];   // (H, H)
  const float* W_out   = (const float*)d_in[10];  // (H, 2H)
  float* out = (float*)d_out;                     // (B,T,H) | h_f | c_f
  (void)in_sizes; (void)n_in; (void)out_size;

  // workspace layout
  char* w = (char*)d_ws;
  size_t off = 0;
  auto alloc = [&](size_t bytes) { char* p = w + off; off += (bytes + 255) & ~size_t(255); return p; };
  unsigned short* Wih_bf    = (unsigned short*)alloc((size_t)4 * H_ * K1_ * 2);
  unsigned short* Whh_bf    = (unsigned short*)alloc((size_t)4 * H_ * H_ * 2);
  unsigned short* Win_bf    = (unsigned short*)alloc((size_t)H_ * H_ * 2);
  unsigned short* Wout_bf   = (unsigned short*)alloc((size_t)H_ * 2 * H_ * 2);
  unsigned short* emb_bf    = (unsigned short*)alloc((size_t)B_ * T_ * E_ * 2);
  unsigned short* htilde_bf = (unsigned short*)alloc((size_t)B_ * H_ * 2);
  unsigned short* concat0   = (unsigned short*)alloc((size_t)B_ * 2048 * 2);
  unsigned short* concat1   = (unsigned short*)alloc((size_t)B_ * 2048 * 2);
  float* h_st     = (float*)alloc((size_t)B_ * H_ * 4);
  float* c_st     = (float*)alloc((size_t)B_ * H_ * 4);
  float* target_f = (float*)alloc((size_t)B_ * H_ * 4);
  // optional bf16 ctx copy (268 MB) — halves the dominant HBM stream
  const size_t ctx_elems = (size_t)B_ * S_ * H_;
  unsigned short* ctx_bf = nullptr;
  if (ws_size >= off + ctx_elems * 2 + 256)
    ctx_bf = (unsigned short*)alloc(ctx_elems * 2);

  auto cvt = [&](const float* src, unsigned short* dst, size_t n) {
    int n4 = (int)(n / 4);
    f32x4_to_bf16x4_kernel<<<(n4 + 255) / 256, 256, 0, stream>>>(
        (const float4*)src, (uint2*)dst, n4);
  };
  cvt(W_ih,    Wih_bf,  (size_t)4 * H_ * K1_);
  cvt(W_hh,    Whh_bf,  (size_t)4 * H_ * H_);
  cvt(W_in,    Win_bf,  (size_t)H_ * H_);
  cvt(W_out,   Wout_bf, (size_t)H_ * 2 * H_);
  cvt(trg_emb, emb_bf,  (size_t)B_ * T_ * E_);
  if (ctx_bf) cvt(ctx, ctx_bf, ctx_elems);
  init_state_kernel<<<(B_ * H_ + 255) / 256, 256, 0, stream>>>(
      h0, c0, h_st, c_st, htilde_bf, concat0);

  unsigned short* concat[2] = {concat0, concat1};
  for (int t = 0; t < T_; ++t) {
    unsigned short* cprev = concat[t & 1];
    unsigned short* cnext = concat[(t + 1) & 1];

    // gates + LSTM pointwise -> h_st, c_st, h(bf16) into cnext[:,1024:]
    gates_lstm_kernel<<<dim3(B_ / 16, H_ / 64), 128, 0, stream>>>(
        emb_bf, t, htilde_bf, cprev, Wih_bf, Whh_bf, b_ih, b_hh,
        c_st, h_st, cnext);

    // target = h @ W_in^T  (A = h bf16 in cnext[:,1024:], lda=2048)
    gemm_tile_kernel<2048, H_, 0><<<dim3(B_ / 16, H_ / 64), 128, 0, stream>>>(
        cnext + 1024, Win_bf, target_f, H_, nullptr);

    // attention: weighted (bf16) -> cnext[:,0:1024]
    if (ctx_bf)
      attn_bf16_kernel<<<B_, 128, 0, stream>>>(ctx_bf, target_f, src_len, cnext);
    else
      attn_f32_kernel<<<B_, 256, 0, stream>>>(ctx, target_f, src_len, cnext);

    // h_tilde = tanh([weighted|h] @ W_out^T) -> d_out[:,t,:] + htilde_bf
    gemm_tile_kernel<2048, 2 * H_, 1><<<dim3(B_ / 16, H_ / 64), 128, 0, stream>>>(
        cnext, Wout_bf, out + (size_t)t * H_, T_ * H_, htilde_bf);
  }

  finalize_kernel<<<(B_ * H_ + 255) / 256, 256, 0, stream>>>(
      h_st, c_st, out + (size_t)B_ * T_ * H_);
}
